// IRGModel_31147102831072
// MI455X (gfx1250) — matrix operand order repaired and checked by reading, not compiled, not run on hardware
//
#include <hip/hip_runtime.h>
#include <math.h>

typedef __bf16 v16bf __attribute__((ext_vector_type(16)));
typedef float  v8f   __attribute__((ext_vector_type(8)));
typedef int    v4i_  __attribute__((ext_vector_type(4)));

// Problem constants (from reference)
constexpr int BATCH = 2, L1 = 4096, L2 = 1024;
constexpr int M1 = 3072, M2 = 1024, E = 2048, H = 16, HD = 128;
constexpr float QSCALE = 0.08838834764831845f;  // HD^-0.5

// ---------------------------------------------------------------- async copy helpers
#if __has_builtin(__builtin_amdgcn_global_load_async_to_lds_b128) && \
    __has_builtin(__builtin_amdgcn_s_wait_asynccnt)
#define ASYNC_OK 1
#else
#define ASYNC_OK 0
#endif

__device__ __forceinline__ void cp16_sync(void* dst, const void* src) {
  *(uint4*)dst = *(const uint4*)src;
}

#if ASYNC_OK
__device__ __forceinline__ void cp16_async(void* lds, const void* glob) {
  __builtin_amdgcn_global_load_async_to_lds_b128(
      (__attribute__((address_space(1))) v4i_*)glob,
      (__attribute__((address_space(3))) v4i_*)lds, 0, 0);
}
#define ASYNC_WAIT(n) __builtin_amdgcn_s_wait_asynccnt(n)
#else
#define cp16_async cp16_sync
#define ASYNC_WAIT(n)
#endif

// ---------------------------------------------------------------- conversion (8 f32 -> 8 bf16 per thread)
__global__ void cvt_f32_bf16(const float* __restrict__ s, __bf16* __restrict__ d, int n8) {
  int i = blockIdx.x * blockDim.x + threadIdx.x;
  if (i >= n8) return;
  const float4* s4 = (const float4*)s + (size_t)i * 2;
  float4 a = s4[0], b = s4[1];
  union { uint4 q; __bf16 e[8]; } o;
  o.e[0] = (__bf16)a.x; o.e[1] = (__bf16)a.y; o.e[2] = (__bf16)a.z; o.e[3] = (__bf16)a.w;
  o.e[4] = (__bf16)b.x; o.e[5] = (__bf16)b.y; o.e[6] = (__bf16)b.z; o.e[7] = (__bf16)b.w;
  *((uint4*)d + i) = o.q;
}

// ---------------------------------------------------------------- GEMM: Y = X * W^T + b (optionally *scale)
// LDS-tiled, double-buffered async staging.  Block = 8 waves; block tile
// 128 rows x 64 cols; wave computes 16x64; k-step 32.
template <int STORE_BF16>
__global__ __launch_bounds__(256) void gemm_xwt(
    const __bf16* __restrict__ X, const __bf16* __restrict__ W,
    const float* __restrict__ bias, float scale,
    float* __restrict__ Yf, __bf16* __restrict__ Yb,
    int N, int K, int Nout) {
  __shared__ __bf16 As[2][128][32];   // 16 KB
  __shared__ __bf16 Bs[2][64][32];    //  8 KB

  const int t    = threadIdx.x;
  const int lane = t & 31;
  const int wave = t >> 5;
  const int m    = lane & 15;
  const int aoff = (lane < 16) ? 0 : 8;    // A frag K-half select
  const int boff = (lane < 16) ? 0 : 8;    // B frag K-half select
  const int rowg = blockIdx.y * 128;
  const int col0 = blockIdx.x * 64;

  // staging coords: A 128x32 (2 threads/row, 32B each), B 64x32 (4 threads/row, 16B each)
  const int ar = t >> 1, ac = (t & 1) * 16;
  const int br = t >> 2, bc = (t & 3) * 8;
  const __bf16* Ag = X + (size_t)(rowg + ar) * K + ac;
  const __bf16* Bg = W + (size_t)(col0 + br) * K + bc;

  auto stage = [&](int buf, int k0) {
    cp16_async(&As[buf][ar][ac],     Ag + k0);
    cp16_async(&As[buf][ar][ac + 8], Ag + k0 + 8);
    cp16_async(&Bs[buf][br][bc],     Bg + k0);
  };

  v8f acc[4] = {};
  const int nk = K / 32;
  stage(0, 0);
  for (int s = 0; s < nk; ++s) {
    __syncthreads();                       // prev compute done before re-staging its buffer
    if (s + 1 < nk) {
      stage((s + 1) & 1, (s + 1) * 32);    // prefetch next tile (async)
      ASYNC_WAIT(3);                       // 3 newest may be pending; older (cur buf) complete
    } else {
      ASYNC_WAIT(0);
    }
    __syncthreads();

    const int buf = s & 1;
    v16bf a;
    {
      const __bf16* ap = &As[buf][wave * 16 + m][aoff];
      ((uint4*)&a)[0] = *(const uint4*)ap;
      ((uint4*)&a)[1] = *(const uint4*)(ap + 16);
    }
#pragma unroll
    for (int c = 0; c < 4; ++c) {
      v16bf bm;
      const __bf16* bp = &Bs[buf][c * 16 + m][boff];
      ((uint4*)&bm)[0] = *(const uint4*)bp;
      ((uint4*)&bm)[1] = *(const uint4*)(bp + 16);
      acc[c] = __builtin_amdgcn_wmma_f32_16x16x32_bf16(false, a, false, bm,
                                                       (short)0, acc[c], false, false);
    }
  }

  const int row0 = rowg + wave * 16;
  const int col  = lane & 15;
  const int rb   = (lane < 16) ? 0 : 8;
#pragma unroll
  for (int c = 0; c < 4; ++c) {
    int cc = col0 + c * 16 + col;
    float bz = bias[cc];
#pragma unroll
    for (int i = 0; i < 8; ++i) {
      float v = (acc[c][i] + bz) * scale;
      size_t idx = (size_t)(row0 + i + rb) * Nout + cc;
      if (STORE_BF16) Yb[idx] = (__bf16)v; else Yf[idx] = v;
    }
  }
}

// ---------------------------------------------------------------- flash attention
// One wave = 16 query rows of one (b,h); 8 waves/block share 32-key K/V chunks.
__global__ __launch_bounds__(256) void flash_attn(
    const __bf16* __restrict__ Q, const __bf16* __restrict__ Kg,
    const __bf16* __restrict__ V, __bf16* __restrict__ O,
    int Lq, int Lk) {
  __shared__ __bf16 Kl[32][HD];        // K chunk (key, dim)       8 KB
  __shared__ __bf16 Vt[HD][32];        // V chunk transposed        8 KB
  __shared__ __bf16 Pl[8][16][32];     // per-wave P re-layout      8 KB

  const int t    = threadIdx.x;
  const int lane = t & 31;
  const int wave = t >> 5;
  const int h = blockIdx.y, b = blockIdx.z;
  const int q0 = (blockIdx.x * 8 + wave) * 16;

  const __bf16* Qb = Q  + (size_t)b * Lq * E + (size_t)h * HD;
  const __bf16* Kb = Kg + (size_t)b * Lk * E + (size_t)h * HD;
  const __bf16* Vb = V  + (size_t)b * Lk * E + (size_t)h * HD;

  const int mrow = lane & 15;
  const int aoff = (lane < 16) ? 0 : 8;
  const int boff = (lane < 16) ? 0 : 8;

  // Preload Q fragments (16 x 128) -- loop invariant, b128 loads.
  v16bf aq[4];
  {
    const __bf16* qr = Qb + (size_t)(q0 + mrow) * E;
#pragma unroll
    for (int d = 0; d < 4; ++d) {
      ((uint4*)&aq[d])[0] = *(const uint4*)(qr + d * 32 + aoff);
      ((uint4*)&aq[d])[1] = *(const uint4*)(qr + d * 32 + 16 + aoff);
    }
  }

  v8f o[8] = {};
  float mrun[8], lrun[8];
#pragma unroll
  for (int i = 0; i < 8; ++i) { mrun[i] = -__builtin_inff(); lrun[i] = 0.0f; }

  for (int k0 = 0; k0 < Lk; k0 += 32) {
    __syncthreads();
    {  // stage chunk: K via async-to-LDS, V via vector load + transpose store
      const int r  = t >> 3;
      const int c0 = (t & 7) * 16;
      const __bf16* kr = Kb + (size_t)(k0 + r) * E + c0;
      const __bf16* vr = Vb + (size_t)(k0 + r) * E + c0;
      cp16_async(&Kl[r][c0],     kr);
      cp16_async(&Kl[r][c0 + 8], kr + 8);
      union { uint4 q[2]; __bf16 e[16]; } vt;
      vt.q[0] = *(const uint4*)vr;
      vt.q[1] = *(const uint4*)(vr + 8);
#pragma unroll
      for (int i = 0; i < 16; ++i) Vt[c0 + i][r] = vt.e[i];
      if (k0 + 32 < Lk) {
        __builtin_prefetch(kr + (size_t)32 * E, 0, 0);
        __builtin_prefetch(vr + (size_t)32 * E, 0, 0);
      }
    }
    ASYNC_WAIT(0);
    __syncthreads();

    // S (16x32) = Q (16x128) * K^T
    v8f s0 = {}, s1 = {};
#pragma unroll
    for (int d = 0; d < 4; ++d) {
      v16bf b0, b1;
      const __bf16* p0 = &Kl[mrow][d * 32 + boff];
      const __bf16* p1 = &Kl[mrow + 16][d * 32 + boff];
      ((uint4*)&b0)[0] = *(const uint4*)p0;
      ((uint4*)&b0)[1] = *(const uint4*)(p0 + 16);
      ((uint4*)&b1)[0] = *(const uint4*)p1;
      ((uint4*)&b1)[1] = *(const uint4*)(p1 + 16);
      s0 = __builtin_amdgcn_wmma_f32_16x16x32_bf16(false, aq[d], false, b0, (short)0, s0, false, false);
      s1 = __builtin_amdgcn_wmma_f32_16x16x32_bf16(false, aq[d], false, b1, (short)0, s1, false, false);
    }

    // Online softmax; each C-row lives in one 16-lane half (xor 1..8 stays inside).
#pragma unroll
    for (int i = 0; i < 8; ++i) {
      float vmax = fmaxf(s0[i], s1[i]);
#pragma unroll
      for (int off = 1; off < 16; off <<= 1)
        vmax = fmaxf(vmax, __shfl_xor(vmax, off, 32));
      float mn = fmaxf(mrun[i], vmax);
      float al = __expf(mrun[i] - mn);
      mrun[i] = mn;
#pragma unroll
      for (int j = 0; j < 8; ++j) o[j][i] *= al;
      s0[i] = __expf(s0[i] - mn);
      s1[i] = __expf(s1[i] - mn);
      float rs = s0[i] + s1[i];
#pragma unroll
      for (int off = 1; off < 16; off <<= 1)
        rs += __shfl_xor(rs, off, 32);
      lrun[i] = lrun[i] * al + rs;
    }

    // Re-layout P (C-layout -> A-fragment) via per-wave LDS tile.
    {
      const int rb = (lane < 16) ? 0 : 8;
      const int c  = lane & 15;
#pragma unroll
      for (int i = 0; i < 8; ++i) {
        Pl[wave][i + rb][c]      = (__bf16)s0[i];
        Pl[wave][i + rb][16 + c] = (__bf16)s1[i];
      }
    }
    __asm__ volatile("" ::: "memory");  // LDS is in-order per wave; just block reordering
    v16bf pa;
    {
      const __bf16* pp = &Pl[wave][mrow][0];
      ((uint4*)&pa)[0] = *(const uint4*)(pp + aoff);
      ((uint4*)&pa)[1] = *(const uint4*)(pp + 16 + aoff);
    }

    // O (16x128) += P (16x32) * V (32x128)
#pragma unroll
    for (int j = 0; j < 8; ++j) {
      v16bf bv;
      const __bf16* vp = &Vt[j * 16 + mrow][boff];
      ((uint4*)&bv)[0] = *(const uint4*)vp;
      ((uint4*)&bv)[1] = *(const uint4*)(vp + 16);
      o[j] = __builtin_amdgcn_wmma_f32_16x16x32_bf16(false, pa, false, bv, (short)0, o[j], false, false);
    }
  }

  {  // normalize and store bf16
    __bf16* Ob = O + (size_t)b * Lq * E + (size_t)h * HD;
    const int rb = (lane < 16) ? 0 : 8;
    const int c  = lane & 15;
#pragma unroll
    for (int j = 0; j < 8; ++j)
#pragma unroll
      for (int i = 0; i < 8; ++i) {
        float v = o[j][i] / lrun[i];
        Ob[(size_t)(q0 + i + rb) * E + j * 16 + c] = (__bf16)v;
      }
  }
}

// ---------------------------------------------------------------- launch
extern "C" void kernel_launch(void* const* d_in, const int* in_sizes, int n_in,
                              void* d_out, int out_size, void* d_ws, size_t ws_size,
                              hipStream_t stream) {
  (void)in_sizes; (void)n_in; (void)out_size; (void)ws_size;
  const float* x1  = (const float*)d_in[0];
  const float* x2  = (const float*)d_in[1];
  const float* Wq  = (const float*)d_in[2];  const float* bq  = (const float*)d_in[3];
  const float* Wk  = (const float*)d_in[4];  const float* bk  = (const float*)d_in[5];
  const float* Wv1 = (const float*)d_in[6];  const float* bv1 = (const float*)d_in[7];
  const float* Wv2 = (const float*)d_in[8];  const float* bv2 = (const float*)d_in[9];
  const float* Wo1 = (const float*)d_in[10]; const float* bo1 = (const float*)d_in[11];
  const float* Wo2 = (const float*)d_in[12]; const float* bo2 = (const float*)d_in[13];

  float* out1 = (float*)d_out;
  float* out2 = out1 + (size_t)BATCH * L1 * M1;

  const size_t N1 = (size_t)BATCH * L1;  // 8192
  const size_t N2 = (size_t)BATCH * L2;  // 2048

  // Bump-allocate bf16 scratch out of d_ws (~230 MB total).
  __bf16* p = (__bf16*)d_ws;
  auto alloc = [&](size_t n) { __bf16* r = p; p += n; return r; };
  __bf16* x1b  = alloc(N1 * M1);
  __bf16* x2b  = alloc(N2 * M2);
  __bf16* Wqb  = alloc((size_t)E * M1);
  __bf16* Wkb  = alloc((size_t)E * M2);
  __bf16* Wv1b = alloc((size_t)E * M1);
  __bf16* Wv2b = alloc((size_t)E * M2);
  __bf16* Wo1b = alloc((size_t)M1 * E);
  __bf16* Wo2b = alloc((size_t)M2 * E);
  __bf16* qb   = alloc(N1 * E);
  __bf16* kb   = alloc(N2 * E);
  __bf16* v1b  = alloc(N1 * E);
  __bf16* v2b  = alloc(N2 * E);
  __bf16* o1b  = alloc(N1 * E);
  __bf16* o2b  = alloc(N2 * E);

  auto cvt = [&](const float* s, __bf16* d, size_t n) {
    unsigned n8 = (unsigned)(n / 8);
    cvt_f32_bf16<<<(n8 + 255) / 256, 256, 0, stream>>>(s, d, (int)n8);
  };
  cvt(x1, x1b, N1 * M1);
  cvt(x2, x2b, N2 * M2);
  cvt(Wq,  Wqb,  (size_t)E * M1);
  cvt(Wk,  Wkb,  (size_t)E * M2);
  cvt(Wv1, Wv1b, (size_t)E * M1);
  cvt(Wv2, Wv2b, (size_t)E * M2);
  cvt(Wo1, Wo1b, (size_t)M1 * E);
  cvt(Wo2, Wo2b, (size_t)M2 * E);

  dim3 blk(256);
  // Projections (softmax scale folded into q)
  gemm_xwt<1><<<dim3(E / 64, (unsigned)(N1 / 128)), blk, 0, stream>>>(x1b, Wqb,  bq,  QSCALE, nullptr, qb,  (int)N1, M1, E);
  gemm_xwt<1><<<dim3(E / 64, (unsigned)(N2 / 128)), blk, 0, stream>>>(x2b, Wkb,  bk,  1.0f,   nullptr, kb,  (int)N2, M2, E);
  gemm_xwt<1><<<dim3(E / 64, (unsigned)(N1 / 128)), blk, 0, stream>>>(x1b, Wv1b, bv1, 1.0f,   nullptr, v1b, (int)N1, M1, E);
  gemm_xwt<1><<<dim3(E / 64, (unsigned)(N2 / 128)), blk, 0, stream>>>(x2b, Wv2b, bv2, 1.0f,   nullptr, v2b, (int)N2, M2, E);

  // p1 = softmax_k(q k^T) -> o1 = p1 @ v2 ; p2 = softmax_q((q k^T)^T) -> o2 = p2 @ v1
  flash_attn<<<dim3(L1 / 128, H, BATCH), blk, 0, stream>>>(qb, kb, v2b, o1b, L1, L2);
  flash_attn<<<dim3(L2 / 128, H, BATCH), blk, 0, stream>>>(kb, qb, v1b, o2b, L2, L1);

  // Output projections (f32 out)
  gemm_xwt<0><<<dim3(M1 / 64, (unsigned)(N1 / 128)), blk, 0, stream>>>(o1b, Wo1b, bo1, 1.0f, out1, nullptr, (int)N1, E, M1);
  gemm_xwt<0><<<dim3(M2 / 64, (unsigned)(N2 / 128)), blk, 0, stream>>>(o2b, Wo2b, bo2, 1.0f, out2, nullptr, (int)N2, E, M2);
}
